// SheafAttentionModule_16870631539372
// MI455X (gfx1250) — matrix operand order repaired and checked by reading, not compiled, not run on hardware
//
#include <hip/hip_runtime.h>
#include <math.h>

typedef __attribute__((ext_vector_type(16))) _Float16 v16h;
typedef __attribute__((ext_vector_type(8)))  _Float16 v8h;
typedef __attribute__((ext_vector_type(8)))  float    v8f;

constexpr int BATCH = 2;
constexpr int SEQ   = 2048;
constexpr int DM    = 512;
constexpr int NH    = 8;
constexpr int DHD   = 64;
constexpr int NP    = 28;           // H*(H-1)/2
constexpr float THRESH_V = 0.1f;
constexpr int ROWS  = BATCH * SEQ;  // 4096

#define DEV __device__ __forceinline__

DEV v8f wmma_f16(v16h a, v16h b, v8f c) {
    // 8 args: (neg_a, A, neg_b, B, c_mod, C, reuse_a, reuse_b)
    return __builtin_amdgcn_wmma_f32_16x16x32_f16(
        false, a, false, b, (short)0, c, false, false);
}

// A-fragment (16x32 f16): lane L -> row M = L&15; halves 0..15 = K(koff..koff+15),
// koff = 16*(L>=16).  base = &A[row][kb].
DEV v16h load_a_frag(const _Float16* base, int koff) {
    v8h lo = *reinterpret_cast<const v8h*>(base + koff);
    v8h hi = *reinterpret_cast<const v8h*>(base + koff + 8);
    return __builtin_shufflevector(lo, hi,
        0,1,2,3,4,5,6,7,8,9,10,11,12,13,14,15);
}

// ---- v_permlane16_b32 16-lane butterfly reductions (VALU, no LDS) ----
DEV float permf16(float v, unsigned s0, unsigned s1) {
    unsigned r = __builtin_amdgcn_permlane16(
        __float_as_uint(v), __float_as_uint(v), s0, s1, false, false);
    return __uint_as_float(r);
}
DEV float rowmax16(float v) {
    v = fmaxf(v, permf16(v, 0x67452301u, 0xEFCDAB89u)); // xor 1
    v = fmaxf(v, permf16(v, 0x45670123u, 0xCDEF89ABu)); // xor 2
    v = fmaxf(v, permf16(v, 0x01234567u, 0x89ABCDEFu)); // xor 4
    v = fmaxf(v, permf16(v, 0xFEDCBA98u, 0x76543210u)); // xor 8
    return v;
}
DEV float rowsum16(float v) {
    v += permf16(v, 0x67452301u, 0xEFCDAB89u);
    v += permf16(v, 0x45670123u, 0xCDEF89ABu);
    v += permf16(v, 0x01234567u, 0x89ABCDEFu);
    v += permf16(v, 0xFEDCBA98u, 0x76543210u);
    return v;
}

// ---- async global -> LDS 16B copy (ASYNCcnt-tracked DMA), via ISA asm ----
DEV void async_copy16(const _Float16* g, _Float16* l) {
    unsigned loff = (unsigned)(size_t)(void*)l;
    asm volatile("global_load_async_to_lds_b128 %0, %1, off"
                 :: "v"(loff), "v"(g)
                 : "memory");
}
DEV void wait_async0() {
    asm volatile("s_wait_asynccnt 0x0" ::: "memory");
}

// ---------------------------------------------------------------- converts
__global__ void f32_to_f16_kernel(const float* __restrict__ src,
                                  _Float16* __restrict__ dst, int n, float scale) {
    int i = blockIdx.x * blockDim.x + threadIdx.x;
    if (i < n) dst[i] = (_Float16)(src[i] * scale);
}

__global__ void zero_f32_kernel(float* p, int n) {
    int i = blockIdx.x * blockDim.x + threadIdx.x;
    if (i < n) p[i] = 0.0f;
}

// ------------------------------------------------- generic WMMA f16 GEMM
// out(row,col) = sum_k A[row][k] * Bw[col][k] + bias[col]*bscale
// MODE 0: half out, [b,h,n,d] (col = h*64+d, row = b*SEQ+n)
// MODE 1: half out, [b,h,d,n] (V transposed)
// MODE 2: float out, row-major stride DM
template <typename OutT, int MODE>
__global__ void gemm_wmma_kernel(const _Float16* __restrict__ A, int lda,
                                 const _Float16* __restrict__ Bw, int ldb,
                                 const float* __restrict__ bias, float bscale,
                                 OutT* __restrict__ out,
                                 int K, int col_tiles) {
    const int lane  = threadIdx.x & 31;
    const int wid   = blockIdx.x * (blockDim.x >> 5) + (threadIdx.x >> 5);
    const int rt    = wid / col_tiles;
    const int ct    = wid % col_tiles;
    const int nlo   = lane & 15;
    const int koffA = (lane >> 4) * 16;
    const int koffB = (lane >> 4) * 16;

    const _Float16* Arow = A  + (size_t)(rt * 16 + nlo) * lda;
    const _Float16* Brow = Bw + (size_t)(ct * 16 + nlo) * ldb;

    v8f acc;
#pragma unroll
    for (int r = 0; r < 8; ++r) acc[r] = 0.0f;

    for (int kb = 0; kb < K; kb += 32) {
        v16h a = load_a_frag(Arow + kb, koffA);
        v16h b = *reinterpret_cast<const v16h*>(Brow + kb + koffB);
        acc = wmma_f16(a, b, acc);
    }

    const int col = ct * 16 + nlo;           // this lane's output column
    const float bv = bias[col] * bscale;
#pragma unroll
    for (int r = 0; r < 8; ++r) {
        int row = rt * 16 + r + ((lane >> 4) << 3);
        float val = acc[r] + bv;
        if constexpr (MODE == 2) {
            out[(size_t)row * DM + col] = (OutT)val;
        } else {
            int b_ = row >> 11;              // SEQ = 2048
            int n  = row & (SEQ - 1);
            int h  = col >> 6;               // DHD = 64
            int d  = col & 63;
            size_t addr;
            if constexpr (MODE == 0)
                addr = (((size_t)(b_ * NH + h)) * SEQ + n) * DHD + d;
            else
                addr = (((size_t)(b_ * NH + h)) * DHD + d) * SEQ + n;
            out[addr] = (OutT)val;
        }
    }
}

// --------------------------------------------------------- flash attention
// 8 waves/block share one (b,h); each wave owns 16 query rows.
// 64-key blocks staged global->LDS with async DMA, double-buffered:
// one butterfly/rescale/barrier per 64 keys. Score scale folded into Wq/bq.
// Writes head_out in f32 (consensus/agg) and f16 (pair WMMA A-matrix).
__global__ void flash_attn_kernel(const _Float16* __restrict__ Qh,
                                  const _Float16* __restrict__ Kh,
                                  const _Float16* __restrict__ Vt,
                                  float* __restrict__ head_out,
                                  _Float16* __restrict__ head_out_h) {
    __shared__ alignas(32) _Float16 ldsP[8][16][64];      // 16 KB P staging
    __shared__ alignas(32) _Float16 ldsK[2][64 * DHD];    // 2 x 8 KB
    __shared__ alignas(32) _Float16 ldsV[2][DHD * 64];    // 2 x 8 KB

    const int tid  = threadIdx.x;
    const int wave = tid >> 5;
    const int lane = tid & 31;
    const int wid  = blockIdx.x * 8 + wave;
    const int qtiles = SEQ / 16;
    const int bh = wid / qtiles;
    const int qt = wid % qtiles;

    const _Float16* Qb = Qh + (size_t)bh * SEQ * DHD;
    const _Float16* Kb = Kh + (size_t)bh * SEQ * DHD;
    const _Float16* Vb = Vt + (size_t)bh * DHD * SEQ;

    const int nlo   = lane & 15;
    const int koffA = (lane >> 4) * 16;
    const int koffB = (lane >> 4) * 16;

    // Q A-fragments over d-chunks [0,32) and [32,64)
    v16h aq[2];
#pragma unroll
    for (int c = 0; c < 2; ++c)
        aq[c] = load_a_frag(Qb + (size_t)(qt * 16 + nlo) * DHD + c * 32, koffA);

    v8f oacc[4];
#pragma unroll
    for (int t = 0; t < 4; ++t)
#pragma unroll
        for (int r = 0; r < 8; ++r) oacc[t][r] = 0.0f;

    float mrun[8], lrun[8];
#pragma unroll
    for (int r = 0; r < 8; ++r) { mrun[r] = -1e30f; lrun[r] = 0.0f; }

    // stage: K tile is one contiguous 8 KB block; V tile = 64 rows x 128 B
    auto stage = [&](int jb, int buf) {
#pragma unroll
        for (int it = 0; it < 2; ++it) {
            int ch = tid + it * 256;                  // 512 chunks of 16 B
            async_copy16(Kb + (size_t)jb * DHD + ch * 8, &ldsK[buf][ch * 8]);
            int d = ch >> 3, c = ch & 7;
            async_copy16(Vb + (size_t)d * SEQ + jb + c * 8,
                         &ldsV[buf][d * 64 + c * 8]);
        }
    };

    stage(0, 0);
    int cur = 0;
    for (int jb = 0; jb < SEQ; jb += 64, cur ^= 1) {
        wait_async0();
        __syncthreads();                     // tile[cur] ready; buffers free

        if (jb + 64 < SEQ) stage(jb + 64, cur ^ 1);   // overlap next DMA

        // S tiles for local key groups g*16..g*16+15
        v8f s[4];
#pragma unroll
        for (int g = 0; g < 4; ++g)
#pragma unroll
            for (int r = 0; r < 8; ++r) s[g][r] = 0.0f;
#pragma unroll
        for (int g = 0; g < 4; ++g)
#pragma unroll
            for (int c = 0; c < 2; ++c) {
                v16h bk = *reinterpret_cast<const v16h*>(
                    &ldsK[cur][(g * 16 + nlo) * DHD + c * 32 + koffB]);
                s[g] = wmma_f16(aq[c], bk, s[g]);
            }

        // online softmax over 64 keys; one butterfly + rescale per row
#pragma unroll
        for (int r = 0; r < 8; ++r) {
            float mx = fmaxf(fmaxf(s[0][r], s[1][r]), fmaxf(s[2][r], s[3][r]));
            mx = rowmax16(mx);
            float mnew  = fmaxf(mrun[r], mx);
            float alpha = __expf(mrun[r] - mnew);
            float p0 = __expf(s[0][r] - mnew);
            float p1 = __expf(s[1][r] - mnew);
            float p2 = __expf(s[2][r] - mnew);
            float p3 = __expf(s[3][r] - mnew);
            float rs = rowsum16((p0 + p1) + (p2 + p3));
            lrun[r] = lrun[r] * alpha + rs;
            mrun[r] = mnew;
#pragma unroll
            for (int t = 0; t < 4; ++t) oacc[t][r] *= alpha;
            int m = r + ((lane >> 4) << 3);
            ldsP[wave][m][nlo]      = (_Float16)p0;
            ldsP[wave][m][nlo + 16] = (_Float16)p1;
            ldsP[wave][m][nlo + 32] = (_Float16)p2;
            ldsP[wave][m][nlo + 48] = (_Float16)p3;
        }
        // same-wave DS ops are in-order; wait stores, re-read as A fragments
        asm volatile("s_wait_dscnt 0x0" ::: "memory");

        v16h ap[2];
#pragma unroll
        for (int c = 0; c < 2; ++c)
            ap[c] = load_a_frag(&ldsP[wave][nlo][c * 32], koffA);

        // O += P @ V   (B frag: column d, K = local key, contiguous in ldsV)
#pragma unroll
        for (int t = 0; t < 4; ++t)
#pragma unroll
            for (int c = 0; c < 2; ++c) {
                v16h bv = *reinterpret_cast<const v16h*>(
                    &ldsV[cur][(t * 16 + nlo) * 64 + c * 32 + koffB]);
                oacc[t] = wmma_f16(ap[c], bv, oacc[t]);
            }
    }

#pragma unroll
    for (int r = 0; r < 8; ++r) {
        float inv = 1.0f / lrun[r];
        int n = qt * 16 + r + ((lane >> 4) << 3);
        size_t base = ((size_t)bh * SEQ + n) * DHD;
#pragma unroll
        for (int t = 0; t < 4; ++t) {
            float v = oacc[t][r] * inv;
            head_out[base + t * 16 + nlo]   = v;
            head_out_h[base + t * 16 + nlo] = (_Float16)v;
        }
    }
}

// -------------------------------------------- pair disagreement via WMMA
// wave: 16 rows of one (b,pair).  ri = hi @ Wr[p]^T (8 WMMAs, K=64),
// df = ri - hj in f32, per-row ssq via permlane butterfly, sqrt-mean atomics.
__global__ void pair_wmma_kernel(const _Float16* __restrict__ hoh,
                                 const float* __restrict__ ho,
                                 const _Float16* __restrict__ Wrh,
                                 float* __restrict__ diff_sum) {
    const int lane = threadIdx.x & 31;
    const int wid  = blockIdx.x * 8 + (threadIdx.x >> 5);
    const int rowtiles = SEQ / 16;           // 128
    const int bp = wid / rowtiles;
    const int rt = wid % rowtiles;
    const int b = bp / NP;
    const int p = bp % NP;

    int i = 0, pp = p, rem = NH - 1;
    while (pp >= rem) { pp -= rem; ++i; --rem; }
    const int j = i + 1 + pp;

    const int nlo   = lane & 15;
    const int koffA = (lane >> 4) * 16;
    const int koffB = (lane >> 4) * 16;

    const _Float16* Ai = hoh + (((size_t)(b * NH + i)) * SEQ + rt * 16 + nlo) * DHD;
    v16h ah[2];
#pragma unroll
    for (int c = 0; c < 2; ++c) ah[c] = load_a_frag(Ai + c * 32, koffA);

    const float* Hj = ho + ((size_t)(b * NH + j)) * SEQ * DHD;
    const _Float16* Wp = Wrh + (size_t)p * DHD * DHD;

    float ssq[8];
#pragma unroll
    for (int r = 0; r < 8; ++r) ssq[r] = 0.0f;

#pragma unroll
    for (int ct = 0; ct < 4; ++ct) {
        v8f acc;
#pragma unroll
        for (int r = 0; r < 8; ++r) acc[r] = 0.0f;
#pragma unroll
        for (int c = 0; c < 2; ++c) {
            v16h bw = *reinterpret_cast<const v16h*>(
                Wp + (size_t)(ct * 16 + nlo) * DHD + c * 32 + koffB);
            acc = wmma_f16(ah[c], bw, acc);
        }
        const int e = ct * 16 + nlo;
#pragma unroll
        for (int r = 0; r < 8; ++r) {
            int n = rt * 16 + r + ((lane >> 4) << 3);
            float df = acc[r] - Hj[(size_t)n * DHD + e];
            ssq[r] += df * df;
        }
    }

    float tot = 0.0f;
#pragma unroll
    for (int r = 0; r < 8; ++r) tot += sqrtf(rowsum16(ssq[r]));
    if (nlo == 0)                            // lanes 0 & 16: each half's 8 rows
        atomicAdd(&diff_sum[b * NP + p], tot * (1.0f / SEQ));
}

// ------------------------------------------------- consensus -> weights
__global__ void weights_kernel(const float* __restrict__ diff_sum,
                               const float* __restrict__ scale_p,
                               float* __restrict__ weights) {
    int b = threadIdx.x;
    if (b >= BATCH) return;
    float sc = scale_p[0];
    float cons[NH];
#pragma unroll
    for (int h = 0; h < NH; ++h) cons[h] = 0.0f;
    int p = 0;
    for (int i = 0; i < NH; ++i)
        for (int j = i + 1; j < NH; ++j, ++p) {
            float s = __expf(-diff_sum[b * NP + p] * sc / THRESH_V);
            cons[i] += s;
            cons[j] += s;
        }
    float f[NH], mx = -1e30f;
#pragma unroll
    for (int h = 0; h < NH; ++h) {
        float c  = cons[h] / (float)(NH - 1);
        float fv = (c > THRESH_V) ? c : 0.0f;
        fv += 1e-10f;
        f[h] = fv;
        mx = fmaxf(mx, fv);
    }
    float se = 0.0f;
#pragma unroll
    for (int h = 0; h < NH; ++h) { f[h] = __expf(f[h] - mx); se += f[h]; }
#pragma unroll
    for (int h = 0; h < NH; ++h) weights[b * NH + h] = f[h] / se;
}

// ------------------------------------------- weighted head aggregation
__global__ void agg_kernel(const float* __restrict__ head_out,
                           const float* __restrict__ weights,
                           _Float16* __restrict__ aggh) {
    int idx = blockIdx.x * blockDim.x + threadIdx.x;   // ROWS*DHD
    if (idx >= ROWS * DHD) return;
    int row = idx / DHD, d = idx % DHD;
    int b = row >> 11, n = row & (SEQ - 1);
    float acc = 0.0f;
#pragma unroll
    for (int h = 0; h < NH; ++h)
        acc += head_out[(((size_t)(b * NH + h)) * SEQ + n) * DHD + d] *
               weights[b * NH + h];
    aggh[idx] = (_Float16)acc;
}

// ------------------------------------------ fold tile(agg)@Wo.T: WoSum
__global__ void wosum_kernel(const float* __restrict__ Wo,
                             _Float16* __restrict__ wos) {
    int idx = blockIdx.x * blockDim.x + threadIdx.x;   // DM*DHD
    if (idx >= DM * DHD) return;
    int o = idx / DHD, d = idx % DHD;
    float s = 0.0f;
#pragma unroll
    for (int h = 0; h < NH; ++h) s += Wo[(size_t)o * DM + h * DHD + d];
    wos[idx] = (_Float16)s;
}

// ---------------------------------------------------------------- launch
extern "C" void kernel_launch(void* const* d_in, const int* in_sizes, int n_in,
                              void* d_out, int out_size, void* d_ws, size_t ws_size,
                              hipStream_t stream) {
    (void)in_sizes; (void)n_in; (void)out_size; (void)ws_size;
    const float* x   = (const float*)d_in[0];
    const float* Wq  = (const float*)d_in[1];
    const float* bq  = (const float*)d_in[2];
    const float* Wk  = (const float*)d_in[3];
    const float* bk  = (const float*)d_in[4];
    const float* Wv  = (const float*)d_in[5];
    const float* bv  = (const float*)d_in[6];
    const float* Wo  = (const float*)d_in[7];
    const float* bo  = (const float*)d_in[8];
    const float* Wr  = (const float*)d_in[9];
    const float* asc = (const float*)d_in[10];
    float* out = (float*)d_out;

    char* ws = (char*)d_ws;
    size_t off = 0;
    auto alloc = [&](size_t bytes) -> void* {
        void* p = ws + off;
        off = (off + bytes + 255) & ~(size_t)255;
        return p;
    };
    _Float16* xh   = (_Float16*)alloc((size_t)ROWS * DM * 2);
    _Float16* Wqh  = (_Float16*)alloc((size_t)DM * DM * 2);
    _Float16* Wkh  = (_Float16*)alloc((size_t)DM * DM * 2);
    _Float16* Wvh  = (_Float16*)alloc((size_t)DM * DM * 2);
    _Float16* Qh   = (_Float16*)alloc((size_t)BATCH * NH * SEQ * DHD * 2);
    _Float16* Kh   = (_Float16*)alloc((size_t)BATCH * NH * SEQ * DHD * 2);
    _Float16* Vt   = (_Float16*)alloc((size_t)BATCH * NH * SEQ * DHD * 2);
    float*    ho   = (float*)   alloc((size_t)BATCH * NH * SEQ * DHD * 4);
    _Float16* hoh  = (_Float16*)alloc((size_t)BATCH * NH * SEQ * DHD * 2);
    _Float16* Wrh  = (_Float16*)alloc((size_t)NP * DHD * DHD * 2);
    float*    dsum = (float*)   alloc(256);
    float*    wts  = (float*)   alloc(256);
    _Float16* aggh = (_Float16*)alloc((size_t)ROWS * DHD * 2);
    _Float16* wosh = (_Float16*)alloc((size_t)DM * DHD * 2);

    const float qscale = 0.125f;             // 1/sqrt(DHD), folded into Q

    // 1. converts (score scale folded into Wq)
    f32_to_f16_kernel<<<(ROWS * DM + 255) / 256, 256, 0, stream>>>(x, xh, ROWS * DM, 1.0f);
    f32_to_f16_kernel<<<(DM * DM + 255) / 256, 256, 0, stream>>>(Wq, Wqh, DM * DM, qscale);
    f32_to_f16_kernel<<<(DM * DM + 255) / 256, 256, 0, stream>>>(Wk, Wkh, DM * DM, 1.0f);
    f32_to_f16_kernel<<<(DM * DM + 255) / 256, 256, 0, stream>>>(Wv, Wvh, DM * DM, 1.0f);
    f32_to_f16_kernel<<<(NP * DHD * DHD + 255) / 256, 256, 0, stream>>>(
        Wr, Wrh, NP * DHD * DHD, 1.0f);

    // 2. QKV projections: (4096x512)@(512x512)^T, 256x32 tiles, 8 waves/block
    const int qkv_tiles = (ROWS / 16) * (DM / 16);   // 8192
    gemm_wmma_kernel<_Float16, 0><<<qkv_tiles / 8, 256, 0, stream>>>(
        xh, DM, Wqh, DM, bq, qscale, Qh, DM, DM / 16);
    gemm_wmma_kernel<_Float16, 0><<<qkv_tiles / 8, 256, 0, stream>>>(
        xh, DM, Wkh, DM, bk, 1.0f, Kh, DM, DM / 16);
    gemm_wmma_kernel<_Float16, 1><<<qkv_tiles / 8, 256, 0, stream>>>(
        xh, DM, Wvh, DM, bv, 1.0f, Vt, DM, DM / 16);

    // 3. flash attention: B*H*(SEQ/16) = 2048 waves, 8 waves/block
    flash_attn_kernel<<<(BATCH * NH * (SEQ / 16)) / 8, 256, 0, stream>>>(
        Qh, Kh, Vt, ho, hoh);

    // 4. pair disagreement (WMMA) -> consensus weights
    zero_f32_kernel<<<1, 64, 0, stream>>>(dsum, BATCH * NP);
    pair_wmma_kernel<<<(BATCH * NP * (SEQ / 16)) / 8, 256, 0, stream>>>(
        hoh, ho, Wrh, dsum);
    weights_kernel<<<1, 32, 0, stream>>>(dsum, asc, wts);

    // 5. aggregation + folded output projection
    agg_kernel<<<(ROWS * DHD + 255) / 256, 256, 0, stream>>>(ho, wts, aggh);
    wosum_kernel<<<(DM * DHD + 255) / 256, 256, 0, stream>>>(Wo, wosh);
    gemm_wmma_kernel<float, 2><<<qkv_tiles / 8, 256, 0, stream>>>(
        aggh, DHD, wosh, DHD, bo, 1.0f, out, DHD, DM / 16);
}
